// Detector_53051436040346
// MI455X (gfx1250) — matrix operand order repaired and checked by reading, not compiled, not run on hardware
//
#include <hip/hip_runtime.h>
#include <hip/hip_bf16.h>
#include <math.h>

// ---------------------------------------------------------------------------
// Types for CDNA5 WMMA (wave32, 16x16x32 bf16 -> f32)
// ---------------------------------------------------------------------------
typedef __attribute__((ext_vector_type(16))) __bf16          bf16x16;
typedef __attribute__((ext_vector_type(16))) unsigned short  u16x16;
typedef __attribute__((ext_vector_type(8)))  unsigned short  u16x8;
typedef __attribute__((ext_vector_type(8)))  float           f32x8;
typedef __attribute__((ext_vector_type(4)))  int             v4i;

#define BATCH   8
#define NA_TOT  79560
#define NCLS    20

// gfx1250 async global->LDS copy (ASYNCcnt path), if the toolchain exposes it.
#if defined(__has_builtin)
#if __has_builtin(__builtin_amdgcn_global_load_async_to_lds_b128)
#define HAVE_ASYNC_LDS 1
#endif
#endif

#ifdef HAVE_ASYNC_LDS
typedef __attribute__((address_space(1))) v4i* gptr_v4i;   // global int4*
typedef __attribute__((address_space(3))) v4i* lptr_v4i;   // LDS int4*
__device__ __forceinline__ void wait_asynccnt0() {
#if __has_builtin(__builtin_amdgcn_s_wait_asynccnt)
    __builtin_amdgcn_s_wait_asynccnt(0);
#else
    asm volatile("s_wait_asynccnt 0x0" ::: "memory");
#endif
}
#endif

__device__ __forceinline__ unsigned short f2bf(float f) {
    unsigned u = __builtin_bit_cast(unsigned, f);
    unsigned r = u + 0x7FFFu + ((u >> 16) & 1u);   // round-to-nearest-even
    return (unsigned short)(r >> 16);
}

// one-time f32 -> bf16 weight conversion
__global__ void w2bf_kernel(const float* __restrict__ src,
                            unsigned short* __restrict__ dst, int n) {
    const int i = blockIdx.x * 256 + threadIdx.x;
    if (i < n) dst[i] = f2bf(src[i]);
}

// ---------------------------------------------------------------------------
// Implicit-GEMM convolution via v_wmma_f32_16x16x32_bf16.
// Block = 256 threads = 8 wave32s laid out 4(M) x 2(N).
// Workgroup tile: 128 cout x 64 pixels, K staged 64-deep.
// Each wave: 32(M) x 32(N) -> 8 WMMAs per stage
//   (2 M-subtiles x 2 N-subtiles x 2 K-halves), A/B fragments reused 2x each.
// Weights pre-converted bf16, staged to LDS with async global->LDS b128
// (no VALU conversion). KSZ is a template constant so the im2col index math
// strength-reduces (no runtime division in the hot loop).
// ---------------------------------------------------------------------------
template <int KSZ>
__launch_bounds__(256, 2)
__global__ void conv_wmma_kernel(const float* __restrict__ X,
                                 const unsigned short* __restrict__ Wbf,
                                 const float* __restrict__ Bias,
                                 const float* __restrict__ Res,
                                 float* __restrict__ Y,
                                 int Cin, int H, int Wd,
                                 int Cout, int Hout, int Wout,
                                 int stride, int pad,
                                 int reluOut, int reluIn) {
    constexpr int KSZ2 = KSZ * KSZ;
    __shared__ __align__(16) unsigned short As[128][72];  // weights [cout][k]
    __shared__ __align__(16) unsigned short Bs[64][72];   // im2col  [pixel][k]

    const int b     = blockIdx.z;
    const int coBlk = blockIdx.y * 128;
    const int pxBlk = blockIdx.x * 64;
    const int tid   = threadIdx.x;
    const int lane  = tid & 31;
    const int wid   = tid >> 5;     // 0..7
    const int wm    = wid >> 1;     // 0..3  (32-row M band)
    const int wn    = wid & 1;      // 0..1  (32-col N band)

    const int K     = Cin * KSZ2;           // always a multiple of 64 here
    const int HWout = Hout * Wout;

    // cooperative A (weight) load: row = tid>>1 (0..127), 32 contiguous k
    const int lda_row = tid >> 1;
    const int lda_k   = (tid & 1) * 32;
    const bool aOK    = (coBlk + lda_row) < Cout;
    const unsigned short* wrow = Wbf + (size_t)(coBlk + lda_row) * K;

    // cooperative B (im2col) load: pixel = tid>>2 (0..63), 16 contiguous k
    const int ldb_pix = tid >> 2;
    const int ldb_k   = (tid & 3) * 16;
    const int gpix    = pxBlk + ldb_pix;
    const bool pixOK  = gpix < HWout;
    int oh = 0, ow = 0;
    if (pixOK) { oh = gpix / Wout; ow = gpix - oh * Wout; }
    const int ihBase = oh * stride - pad;
    const int iwBase = ow * stride - pad;
    const float* Xb  = X + (size_t)b * Cin * H * Wd;

    // wave32 fragment coordinates; k = 16*(lane>>4) + s (contiguous) for A and B
    const int arow0 = wm * 32 + (lane & 15);  // M sub-tile 0
    const int arow1 = arow0 + 16;             // M sub-tile 1
    const int abase = (lane >> 4) * 16;       // A: 16 contiguous K {base, base+8} (+ko)
    const int brow0 = wn * 32 + (lane & 15);  // N sub-tile 0
    const int brow1 = brow0 + 16;             // N sub-tile 1
    const int bk    = (lane >> 4) * 16;       // B: 16 contiguous K (+ko)

    f32x8 acc00 = {0.f, 0.f, 0.f, 0.f, 0.f, 0.f, 0.f, 0.f};
    f32x8 acc01 = acc00, acc10 = acc00, acc11 = acc00;

    for (int k0 = 0; k0 < K; k0 += 64) {
        // ---- stage A: bf16 weight tile, straight copy (no conversion) ----
#ifdef HAVE_ASYNC_LDS
        if (aOK) {
            #pragma unroll
            for (int q = 0; q < 4; ++q)
                __builtin_amdgcn_global_load_async_to_lds_b128(
                    (gptr_v4i)(wrow + k0 + lda_k + 8 * q),
                    (lptr_v4i)&As[lda_row][lda_k + 8 * q], 0, 0);
        } else {
            const u16x8 z = {0, 0, 0, 0, 0, 0, 0, 0};
            #pragma unroll
            for (int q = 0; q < 4; ++q)
                *(u16x8*)&As[lda_row][lda_k + 8 * q] = z;
        }
#else
        {
            #pragma unroll
            for (int q = 0; q < 4; ++q) {
                u16x8 w0;
                if (aOK) w0 = *(const u16x8*)(wrow + k0 + lda_k + 8 * q);
                else     w0 = (u16x8){0, 0, 0, 0, 0, 0, 0, 0};
                *(u16x8*)&As[lda_row][lda_k + 8 * q] = w0;
            }
            if (aOK && (k0 + 64) < K)
                __builtin_prefetch(wrow + k0 + 64 + lda_k, 0, 1);
        }
#endif
        // ---- stage B: im2col gather + f32->bf16 (pixel-major: K contiguous) --
        {
            u16x8 t0, t1;
            #pragma unroll
            for (int e = 0; e < 16; ++e) {
                const int k = k0 + ldb_k + e;
                float v = 0.f;
                if (pixOK) {
                    int ci, kh, kw;
                    if constexpr (KSZ == 1) {
                        ci = k; kh = 0; kw = 0;
                    } else {
                        ci = k / KSZ2;
                        const int r = k - ci * KSZ2;
                        kh = r / KSZ;
                        kw = r - kh * KSZ;
                    }
                    const int ih = ihBase + kh, iw = iwBase + kw;
                    if (ih >= 0 && ih < H && iw >= 0 && iw < Wd) {
                        v = Xb[((size_t)ci * H + ih) * Wd + iw];
                        if (reluIn) v = fmaxf(v, 0.f);
                    }
                }
                if (e < 8) t0[e] = f2bf(v); else t1[e - 8] = f2bf(v);
            }
            *(u16x8*)&Bs[ldb_pix][ldb_k]     = t0;
            *(u16x8*)&Bs[ldb_pix][ldb_k + 8] = t1;
        }
#ifdef HAVE_ASYNC_LDS
        wait_asynccnt0();
#endif
        __syncthreads();

        // ---- fragments in contiguous-k order (A and B alike), 8 WMMAs per stage ----
        #pragma unroll
        for (int ko = 0; ko < 64; ko += 32) {
            u16x8 a0lo = *(const u16x8*)&As[arow0][ko + abase];
            u16x8 a0hi = *(const u16x8*)&As[arow0][ko + abase + 8];
            u16x8 a1lo = *(const u16x8*)&As[arow1][ko + abase];
            u16x8 a1hi = *(const u16x8*)&As[arow1][ko + abase + 8];
            bf16x16 a0 = __builtin_bit_cast(bf16x16,
                __builtin_shufflevector(a0lo, a0hi,
                    0, 1, 2, 3, 4, 5, 6, 7, 8, 9, 10, 11, 12, 13, 14, 15));
            bf16x16 a1 = __builtin_bit_cast(bf16x16,
                __builtin_shufflevector(a1lo, a1hi,
                    0, 1, 2, 3, 4, 5, 6, 7, 8, 9, 10, 11, 12, 13, 14, 15));
            bf16x16 b0 = __builtin_bit_cast(bf16x16,
                *(const u16x16*)&Bs[brow0][ko + bk]);
            bf16x16 b1 = __builtin_bit_cast(bf16x16,
                *(const u16x16*)&Bs[brow1][ko + bk]);
            acc00 = __builtin_amdgcn_wmma_f32_16x16x32_bf16(
                        false, a0, false, b0, (short)0, acc00, false, false);
            acc01 = __builtin_amdgcn_wmma_f32_16x16x32_bf16(
                        false, a0, false, b1, (short)0, acc01, false, false);
            acc10 = __builtin_amdgcn_wmma_f32_16x16x32_bf16(
                        false, a1, false, b0, (short)0, acc10, false, false);
            acc11 = __builtin_amdgcn_wmma_f32_16x16x32_bf16(
                        false, a1, false, b1, (short)0, acc11, false, false);
        }
        __syncthreads();
    }

    // ---- epilogue: D layout — VGPR r holds M=r+8*(lane>=16), N=lane&15 ----
    const int n   = lane & 15;
    const int mof = (lane >> 4) * 8;
    auto store_tile = [&](const f32x8& acc, int sm, int sn) {
        const int p = pxBlk + wn * 32 + sn * 16 + n;
        if (p >= HWout) return;
        const int oh2 = p / Wout, ow2 = p - (p / Wout) * Wout;
        #pragma unroll
        for (int r = 0; r < 8; ++r) {
            const int co = coBlk + wm * 32 + sm * 16 + mof + r;
            if (co < Cout) {
                const size_t idx =
                    (((size_t)b * Cout + co) * Hout + oh2) * Wout + ow2;
                float v = acc[r] + Bias[co];
                if (Res) v += Res[idx];
                if (reluOut) v = fmaxf(v, 0.f);
                Y[idx] = v;
            }
        }
    };
    store_tile(acc00, 0, 0);
    store_tile(acc01, 0, 1);
    store_tile(acc10, 1, 0);
    store_tile(acc11, 1, 1);
}

// ---------------------------------------------------------------------------
// 2x-1 interleaved bilinear upsample (exact reference semantics)
// ---------------------------------------------------------------------------
__global__ void upsample_kernel(const float* __restrict__ in,
                                float* __restrict__ out,
                                int C, int H, int W) {
    const int Ho = 2 * H - 1, Wo = 2 * W - 1;
    const long long total = (long long)BATCH * C * Ho * Wo;
    long long i = (long long)blockIdx.x * 256 + threadIdx.x;
    if (i >= total) return;
    const int x = (int)(i % Wo);
    long long t = i / Wo;
    const int y = (int)(t % Ho);
    const long long bc = t / Ho;
    const float* s = in + bc * (long long)H * W;
    const int y0 = y >> 1, x0 = x >> 1;
    float v;
    if (!(y & 1) && !(x & 1))      v = s[y0 * W + x0];
    else if (!(y & 1))             v = 0.5f * (s[y0 * W + x0] + s[y0 * W + x0 + 1]);
    else if (!(x & 1))             v = 0.5f * (s[y0 * W + x0] + s[(y0 + 1) * W + x0]);
    else                           v = 0.25f * (s[y0 * W + x0] + s[y0 * W + x0 + 1] +
                                                s[(y0 + 1) * W + x0] + s[(y0 + 1) * W + x0 + 1]);
    out[i] = v;
}

// ---------------------------------------------------------------------------
// (B, 9*nch, H, W)  ->  (B, NA_TOT, nch) slice  (transpose(0,2,3,1).reshape)
// ---------------------------------------------------------------------------
__global__ void scatter_head_kernel(const float* __restrict__ src,
                                    float* __restrict__ dst,
                                    int H, int W, int nch, int abase) {
    const long long total = (long long)BATCH * H * W * 9 * nch;
    long long i = (long long)blockIdx.x * 256 + threadIdx.x;
    if (i >= total) return;
    const int c = (int)(i % nch); long long t = i / nch;
    const int a = (int)(t % 9);   t /= 9;
    const int w = (int)(t % W);   t /= W;
    const int h = (int)(t % H);   t /= H;
    const int b = (int)t;
    dst[((size_t)b * NA_TOT + abase + ((size_t)h * W + w) * 9 + a) * nch + c] =
        src[(((size_t)b * (9 * nch) + a * nch + c) * H + h) * W + w];
}

// ---------------------------------------------------------------------------
// Anchor generation (analytic) + IoU + encoding + losses
// ---------------------------------------------------------------------------
__constant__ float c_ahw[18] = {32.f, 32.f, 40.32f, 40.32f, 50.8f, 50.8f,
                                22.63f, 45.25f, 28.51f, 57.02f, 35.92f, 71.84f,
                                45.25f, 22.63f, 57.02f, 28.51f, 71.84f, 35.92f};

__device__ __forceinline__ void anchor_at(int a, float& cy, float& cx,
                                          float& h, float& w) {
    const int   offs[6]   = {0, 59049, 74178, 78147, 79236, 79560};
    const int   fs[5]     = {81, 41, 21, 11, 6};
    const float strd[5]   = {8.f, 16.f, 32.f, 64.f, 128.f};
    int s = 0;
    while (s < 4 && a >= offs[s + 1]) ++s;
    const int t = a - offs[s];
    const int f = fs[s];
    const int i = t / (f * 9);
    const int r = t - i * (f * 9);
    const int j = r / 9;
    const int k = r - j * 9;
    cy = (i + 0.5f) * strd[s];
    cx = (j + 0.5f) * strd[s];
    const float sc = (float)(1 << s);
    h = c_ahw[2 * k] * sc;
    w = c_ahw[2 * k + 1] * sc;
}

__device__ __forceinline__ float iou_fn(float ay0, float ax0, float ay1, float ax1,
                                        float by0, float bx0, float by1, float bx1) {
    const float ty = fmaxf(ay0, by0), tx = fmaxf(ax0, bx0);
    const float my = fminf(ay1, by1), mx = fminf(ax1, bx1);
    const float ih = fmaxf(my - ty, 0.f), iw = fmaxf(mx - tx, 0.f);
    const float inter = ih * iw;
    const float aa = (ay1 - ay0) * (ax1 - ax0);
    const float ab = (by1 - by0) * (bx1 - bx0);
    return inter / (aa + ab - inter);
}

__device__ __forceinline__ void box2off(float y0, float x0, float y1, float x1,
                                        float ay, float ax, float ah, float aw,
                                        float* o) {
    const float cy = 0.5f * (y0 + y1), cx = 0.5f * (x0 + x1);
    const float h = y1 - y0, w = x1 - x0;
    o[0] = (cy - ay) / ah;
    o[1] = (cx - ax) / aw;
    o[2] = logf(h / ah + 1e-10f);
    o[3] = logf(w / aw + 1e-10f);
}

__global__ void enc_pass1_kernel(const float* __restrict__ lbox,
                                 int* __restrict__ cls_t, float* __restrict__ reg_t,
                                 int* __restrict__ lab_sel, int* __restrict__ posb) {
    const int a = blockIdx.x * 256 + threadIdx.x;
    const int b = blockIdx.y;
    if (a >= NA_TOT) return;
    float cy, cx, ah, aw;
    anchor_at(a, cy, cx, ah, aw);
    const float ay0 = cy - 0.5f * ah, ax0 = cx - 0.5f * aw;
    const float ay1 = cy + 0.5f * ah, ax1 = cx + 0.5f * aw;
    float best = -1.f; int bj = 0;
    bool anyhi = false, alllo = true;
    const float* bb = lbox + (size_t)b * 20 * 4;
    for (int j = 0; j < 20; ++j) {
        const float v = iou_fn(ay0, ax0, ay1, ax1,
                               bb[4 * j], bb[4 * j + 1], bb[4 * j + 2], bb[4 * j + 3]);
        if (v > best) { best = v; bj = j; }
        anyhi = anyhi || (v > 0.5f);
        alllo = alllo && (v < 0.3f);
    }
    const size_t o = (size_t)b * NA_TOT + a;
    lab_sel[o] = bj;
    posb[o]    = anyhi ? 1 : 0;
    cls_t[o]   = alllo ? 0 : -1;
    reg_t[4 * o] = 0.f; reg_t[4 * o + 1] = 0.f; reg_t[4 * o + 2] = 0.f; reg_t[4 * o + 3] = 0.f;
}

__global__ void enc_box_argmax_kernel(const float* __restrict__ lbox,
                                      int* __restrict__ anch_sel) {
    __shared__ float sV[256];
    __shared__ int   sI[256];
    const int j = blockIdx.x, b = blockIdx.y, tid = threadIdx.x;
    const float* bx = lbox + ((size_t)b * 20 + j) * 4;
    const float by0 = bx[0], bx0 = bx[1], by1 = bx[2], bx1 = bx[3];
    float best = -1.f; int bi = 0x7fffffff;
    for (int a = tid; a < NA_TOT; a += 256) {
        float cy, cx, ah, aw;
        anchor_at(a, cy, cx, ah, aw);
        const float v = iou_fn(cy - 0.5f * ah, cx - 0.5f * aw,
                               cy + 0.5f * ah, cx + 0.5f * aw,
                               by0, bx0, by1, bx1);
        if (v > best) { best = v; bi = a; }
    }
    sV[tid] = best; sI[tid] = bi;
    __syncthreads();
    for (int s = 128; s > 0; s >>= 1) {
        if (tid < s) {
            if (sV[tid + s] > sV[tid] ||
                (sV[tid + s] == sV[tid] && sI[tid + s] < sI[tid])) {
                sV[tid] = sV[tid + s]; sI[tid] = sI[tid + s];
            }
        }
        __syncthreads();
    }
    if (tid == 0) anch_sel[b * 20 + j] = sI[0];
}

__global__ void enc_scatter_kernel(const int* __restrict__ lclass,
                                   const float* __restrict__ lbox,
                                   const int* __restrict__ anch_sel,
                                   int* __restrict__ cls_t, float* __restrict__ reg_t) {
    const int t = blockIdx.x * 256 + threadIdx.x;
    if (t >= BATCH * 20) return;
    const int b = t / 20;
    const int a = anch_sel[t];
    const size_t o = (size_t)b * NA_TOT + a;
    cls_t[o] = lclass[t];
    const float* bx = lbox + (size_t)t * 4;
    float cy, cx, ah, aw;
    anchor_at(a, cy, cx, ah, aw);
    box2off(bx[0], bx[1], bx[2], bx[3], cy, cx, ah, aw, &reg_t[4 * o]);
}

__global__ void enc_override_kernel(const int* __restrict__ lclass,
                                    const float* __restrict__ lbox,
                                    const int* __restrict__ lab_sel,
                                    const int* __restrict__ posb,
                                    int* __restrict__ cls_t, float* __restrict__ reg_t) {
    const int a = blockIdx.x * 256 + threadIdx.x;
    const int b = blockIdx.y;
    if (a >= NA_TOT) return;
    const size_t o = (size_t)b * NA_TOT + a;
    if (!posb[o]) return;
    const int j = lab_sel[o];
    cls_t[o] = lclass[b * 20 + j];
    const float* bx = lbox + ((size_t)b * 20 + j) * 4;
    float cy, cx, ah, aw;
    anchor_at(a, cy, cx, ah, aw);
    box2off(bx[0], bx[1], bx[2], bx[3], cy, cx, ah, aw, &reg_t[4 * o]);
}

__device__ __forceinline__ float log_sig(float x) {
    return (x >= 0.f) ? -log1pf(expf(-x)) : x - log1pf(expf(x));
}

__global__ void zero3_kernel(float* out) {
    if (threadIdx.x < 3) out[threadIdx.x] = 0.f;
}

__global__ void loss_kernel(const float* __restrict__ cls_out,
                            const float* __restrict__ reg_out,
                            const int* __restrict__ cls_t,
                            const float* __restrict__ reg_t,
                            float* __restrict__ out) {
    __shared__ float sC[256], sR[256], sN[256];
    const int tid = threadIdx.x;
    const long long i = (long long)blockIdx.x * 256 + tid;
    float lc = 0.f, lr = 0.f, np = 0.f;
    if (i < (long long)BATCH * NA_TOT) {
        const int t = cls_t[i];
        if (t > -1) {
            const float* co = cls_out + i * NCLS;
            float s = 0.f;
            #pragma unroll
            for (int c = 0; c < NCLS; ++c) {
                const float x = co[c];
                const bool oh = (t == c + 1);
                const float lpt = oh ? log_sig(x) : log_sig(-x);
                const float pt  = expf(lpt);
                const float wgt = oh ? 0.25f : 0.75f;
                const float om  = 1.f - pt;
                s += -wgt * om * om * lpt;
            }
            lc = s;
        }
        if (t > 0) {
            const float* ro = reg_out + i * 4;
            const float* rt = reg_t + i * 4;
            float s = 0.f;
            #pragma unroll
            for (int c = 0; c < 4; ++c) {
                const float d = fabsf(ro[c] - rt[c]);
                s += (d < 1.f) ? 0.5f * d * d : d - 0.5f;
            }
            lr = s; np = 1.f;
        }
    }
    sC[tid] = lc; sR[tid] = lr; sN[tid] = np;
    __syncthreads();
    for (int s = 128; s > 0; s >>= 1) {
        if (tid < s) { sC[tid] += sC[tid + s]; sR[tid] += sR[tid + s]; sN[tid] += sN[tid + s]; }
        __syncthreads();
    }
    if (tid == 0) {
        atomicAdd(&out[0], sC[0]);
        atomicAdd(&out[1], sR[0]);
        atomicAdd(&out[2], sN[0]);
    }
}

// ---------------------------------------------------------------------------
// Host orchestration
// ---------------------------------------------------------------------------
extern "C" void kernel_launch(void* const* d_in, const int* in_sizes, int n_in,
                              void* d_out, int out_size, void* d_ws, size_t ws_size,
                              hipStream_t stream) {
    (void)in_sizes; (void)n_in; (void)out_size; (void)ws_size;

    // inputs: setup_inputs() insertion order; params leaves in dict/list order
    const float* C3     = (const float*)d_in[0];   // 8x512x81x81
    const float* C4     = (const float*)d_in[1];   // 8x1024x41x41
    const float* C5     = (const float*)d_in[2];   // 8x2048x21x21
    const int*   lclass = (const int*)  d_in[3];   // 8x20
    const float* lbox   = (const float*)d_in[4];   // 8x20x4
    const float *prj5w = (const float*)d_in[5],  *prj5b = (const float*)d_in[6];
    const float *prj4w = (const float*)d_in[7],  *prj4b = (const float*)d_in[8];
    const float *prj3w = (const float*)d_in[9],  *prj3b = (const float*)d_in[10];
    const float *cv5w  = (const float*)d_in[11], *cv5b  = (const float*)d_in[12];
    const float *cv4w  = (const float*)d_in[13], *cv4b  = (const float*)d_in[14];
    const float *cv3w  = (const float*)d_in[15], *cv3b  = (const float*)d_in[16];
    const float *o6w   = (const float*)d_in[17], *o6b   = (const float*)d_in[18];
    const float *o7w   = (const float*)d_in[19], *o7b   = (const float*)d_in[20];
    const float *clsw[5], *clsb[5], *regw[5], *regb[5];
    for (int i = 0; i < 5; ++i) {
        clsw[i] = (const float*)d_in[21 + i];
        clsb[i] = (const float*)d_in[26 + i];
        regw[i] = (const float*)d_in[31 + i];
        regb[i] = (const float*)d_in[36 + i];
    }
    float* out = (float*)d_out;   // [loss_cls, loss_reg, num_pos]

    // workspace carve
    size_t off = 0;
    auto carve = [&](size_t bytes) -> void* {
        void* r = (char*)d_ws + off;
        off += (bytes + 255) & ~(size_t)255;
        return r;
    };
    const size_t SZ81 = (size_t)BATCH * 256 * 81 * 81;
    const size_t SZ41 = (size_t)BATCH * 256 * 41 * 41;
    const size_t SZ21 = (size_t)BATCH * 256 * 21 * 21;
    const size_t SZ11 = (size_t)BATCH * 256 * 11 * 11;
    const size_t SZ6  = (size_t)BATCH * 256 * 6 * 6;
    float* tmp0    = (float*)carve(SZ81 * 4);
    float* tmp1    = (float*)carve(SZ81 * 4);
    float* P3      = (float*)carve(SZ81 * 4);
    float* P4u     = (float*)carve(SZ81 * 4);
    float* P4      = (float*)carve(SZ41 * 4);
    float* P5u     = (float*)carve(SZ41 * 4);
    float* P5      = (float*)carve(SZ21 * 4);
    float* P6      = (float*)carve(SZ11 * 4);
    float* P7      = (float*)carve(SZ6  * 4);
    float* headout = (float*)carve((size_t)BATCH * 180 * 81 * 81 * 4);
    float* cls_out = (float*)carve((size_t)BATCH * NA_TOT * NCLS * 4);
    float* reg_out = (float*)carve((size_t)BATCH * NA_TOT * 4 * 4);
    float* reg_t   = (float*)carve((size_t)BATCH * NA_TOT * 4 * 4);
    int*   cls_t   = (int*)  carve((size_t)BATCH * NA_TOT * 4);
    int*   lab_sel = (int*)  carve((size_t)BATCH * NA_TOT * 4);
    int*   posb    = (int*)  carve((size_t)BATCH * NA_TOT * 4);
    int*   anch_sel= (int*)  carve((size_t)BATCH * 20 * 4);

    // one-time f32 -> bf16 weight conversion into workspace
    auto cvt = [&](const float* s, size_t n) -> const unsigned short* {
        unsigned short* d = (unsigned short*)carve(n * 2);
        w2bf_kernel<<<dim3((unsigned)((n + 255) / 256)), 256, 0, stream>>>(s, d, (int)n);
        return d;
    };
    const unsigned short* prj5wb = cvt(prj5w, (size_t)256 * 2048);
    const unsigned short* prj4wb = cvt(prj4w, (size_t)256 * 1024);
    const unsigned short* prj3wb = cvt(prj3w, (size_t)256 * 512);
    const unsigned short* cv5wb  = cvt(cv5w,  (size_t)256 * 256 * 9);
    const unsigned short* cv4wb  = cvt(cv4w,  (size_t)256 * 256 * 9);
    const unsigned short* cv3wb  = cvt(cv3w,  (size_t)256 * 256 * 9);
    const unsigned short* o6wb   = cvt(o6w,   (size_t)256 * 2048 * 9);
    const unsigned short* o7wb   = cvt(o7w,   (size_t)256 * 256 * 9);
    const unsigned short *clswb[5], *regwb[5];
    for (int i = 0; i < 4; ++i) {
        clswb[i] = cvt(clsw[i], (size_t)256 * 256 * 9);
        regwb[i] = cvt(regw[i], (size_t)256 * 256 * 9);
    }
    clswb[4] = cvt(clsw[4], (size_t)180 * 256 * 9);
    regwb[4] = cvt(regw[4], (size_t)36 * 256 * 9);

    auto conv = [&](const float* X, int Cin, int H, int W,
                    const unsigned short* Wt, const float* Bs, int Cout,
                    int ksz, int stride, int pad,
                    float* Y, int Hout, int Wout,
                    const float* Res, int reluOut, int reluIn) {
        dim3 g((Hout * Wout + 63) / 64, (Cout + 127) / 128, BATCH);
        if (ksz == 1)
            conv_wmma_kernel<1><<<g, 256, 0, stream>>>(X, Wt, Bs, Res, Y,
                Cin, H, W, Cout, Hout, Wout, stride, pad, reluOut, reluIn);
        else
            conv_wmma_kernel<3><<<g, 256, 0, stream>>>(X, Wt, Bs, Res, Y,
                Cin, H, W, Cout, Hout, Wout, stride, pad, reluOut, reluIn);
    };
    auto upsample = [&](const float* in, float* o, int H, int W) {
        long long total = (long long)BATCH * 256 * (2 * H - 1) * (2 * W - 1);
        upsample_kernel<<<dim3((unsigned)((total + 255) / 256)), 256, 0, stream>>>(in, o, 256, H, W);
    };

    // ------------------------- FPN -------------------------
    conv(C5, 2048, 21, 21, prj5wb, prj5b, 256, 1, 1, 0, tmp0, 21, 21, nullptr, 0, 0);
    upsample(tmp0, P5u, 21, 21);                                         // 41x41
    conv(tmp0, 256, 21, 21, cv5wb, cv5b, 256, 3, 1, 1, P5, 21, 21, nullptr, 0, 0);
    conv(C4, 1024, 41, 41, prj4wb, prj4b, 256, 1, 1, 0, tmp0, 41, 41, P5u, 0, 0);
    upsample(tmp0, P4u, 41, 41);                                         // 81x81
    conv(tmp0, 256, 41, 41, cv4wb, cv4b, 256, 3, 1, 1, P4, 41, 41, nullptr, 0, 0);
    conv(C3, 512, 81, 81, prj3wb, prj3b, 256, 1, 1, 0, tmp0, 81, 81, P4u, 0, 0);
    conv(tmp0, 256, 81, 81, cv3wb, cv3b, 256, 3, 1, 1, P3, 81, 81, nullptr, 0, 0);
    conv(C5, 2048, 21, 21, o6wb, o6b, 256, 3, 2, 1, P6, 11, 11, nullptr, 0, 0);
    conv(P6, 256, 11, 11, o7wb, o7b, 256, 3, 2, 1, P7, 6, 6, nullptr, 0, 1); // relu on input

    // ------------------------- heads -------------------------
    const float* feats[5] = {P3, P4, P5, P6, P7};
    const int    fh[5]    = {81, 41, 21, 11, 6};
    const int    abase[5] = {0, 59049, 74178, 78147, 79236};
    float* pp[2] = {tmp0, tmp1};

    for (int l = 0; l < 5; ++l) {
        const int h = fh[l];
        // classification head
        const float* cur = feats[l];
        for (int i = 0; i < 4; ++i) {
            conv(cur, 256, h, h, clswb[i], clsb[i], 256, 3, 1, 1, pp[i & 1], h, h, nullptr, 1, 0);
            cur = pp[i & 1];
        }
        conv(cur, 256, h, h, clswb[4], clsb[4], 9 * NCLS, 3, 1, 1, headout, h, h, nullptr, 0, 0);
        {
            long long total = (long long)BATCH * h * h * 9 * NCLS;
            scatter_head_kernel<<<dim3((unsigned)((total + 255) / 256)), 256, 0, stream>>>(
                headout, cls_out, h, h, NCLS, abase[l]);
        }
        // regression head
        cur = feats[l];
        for (int i = 0; i < 4; ++i) {
            conv(cur, 256, h, h, regwb[i], regb[i], 256, 3, 1, 1, pp[i & 1], h, h, nullptr, 1, 0);
            cur = pp[i & 1];
        }
        conv(cur, 256, h, h, regwb[4], regb[4], 36, 3, 1, 1, headout, h, h, nullptr, 0, 0);
        {
            long long total = (long long)BATCH * h * h * 9 * 4;
            scatter_head_kernel<<<dim3((unsigned)((total + 255) / 256)), 256, 0, stream>>>(
                headout, reg_out, h, h, 4, abase[l]);
        }
    }

    // ------------------------- anchor encoding -------------------------
    dim3 ganc((NA_TOT + 255) / 256, BATCH);
    enc_pass1_kernel<<<ganc, 256, 0, stream>>>(lbox, cls_t, reg_t, lab_sel, posb);
    enc_box_argmax_kernel<<<dim3(20, BATCH), 256, 0, stream>>>(lbox, anch_sel);
    enc_scatter_kernel<<<1, 256, 0, stream>>>(lclass, lbox, anch_sel, cls_t, reg_t);
    enc_override_kernel<<<ganc, 256, 0, stream>>>(lclass, lbox, lab_sel, posb, cls_t, reg_t);

    // ------------------------- losses -------------------------
    zero3_kernel<<<1, 32, 0, stream>>>(out);
    const long long ltotal = (long long)BATCH * NA_TOT;
    loss_kernel<<<dim3((unsigned)((ltotal + 255) / 256)), 256, 0, stream>>>(
        cls_out, reg_out, cls_t, reg_t, out);
}
